// GNN_MPNN_Model_34832184771009
// MI455X (gfx1250) — compile-verified, matrix operand order checked by reading, not run on hardware
//
#include <hip/hip_runtime.h>
#include <math.h>

typedef __attribute__((ext_vector_type(16))) _Float16 v16h;
typedef __attribute__((ext_vector_type(8)))  _Float16 v8h;
typedef __attribute__((ext_vector_type(4)))  _Float16 v4h;
typedef __attribute__((ext_vector_type(2)))  _Float16 v2h;
typedef __attribute__((ext_vector_type(8)))  float    v8f;

#define NN   50000
#define DIM  128
#define NE   800000
#define OUTD 32

// ---------------------------------------------------------------------------
// Load a 16x32 f16 tile (row-major, leading dim ld) into the CDNA5 WMMA
// 16-bit A/B per-lane layout (v16h per lane):
//   lanes 0-15  (row r = lane):     K 0..7  in h[0..7],  K 16..23 in h[8..15]
//   lanes 16-31 (row r = lane-16):  K 8..15 in h[0..7],  K 24..31 in h[8..15]
// For B = W^T (C = H*W^T, W row-major [out,in]) column j of B is row j of W,
// so the same loader serves both operands. Two 16B loads per matrix per step.
// ---------------------------------------------------------------------------
__device__ __forceinline__ v16h load_tile_h(const _Float16* __restrict__ base,
                                            int ld, int lane) {
  const int r  = lane & 15;
  const int hi = lane >> 4;
  const _Float16* p = base + (size_t)r * ld + (hi ? 8 : 0);
  v8h lo = *(const v8h*)p;
  v8h hh = *(const v8h*)(p + 16);
  v16h o;
#pragma unroll
  for (int t = 0; t < 8; ++t) { o[t] = lo[t]; o[8 + t] = hh[t]; }
  return o;
}

// One wave -> one 16x16 tile of C = H(16x128) * W^T, K-loop of 4 WMMAs.
__device__ __forceinline__ v8f wmma_tile(const _Float16* __restrict__ Hrow,
                                         const _Float16* __restrict__ Wrow,
                                         int lane) {
  v8f acc = {};
#pragma unroll
  for (int k0 = 0; k0 < DIM; k0 += 32) {
    v16h a = load_tile_h(Hrow + k0, DIM, lane);
    v16h b = load_tile_h(Wrow + k0, DIM, lane);
    acc = __builtin_amdgcn_wmma_f32_16x16x32_f16(
        false, a, false, b, (short)0, acc, false, false);
  }
  return acc;
}

// ---------------------------------------------------------------------------
// Fused per-layer GEMM: 16 column-tiles per 16-row tile.
//   ct 0..7  : m   = H @ Wm^T        -> m_out  (f16, feeds the edge gather)
//   ct 8..15 : agg = H @ Wu^T + bu   -> agg_out (f32, scatter accumulates on top)
// C/D f32 layout: VGPR r -> (M = r + 8*(lane>=16), N = lane&15).
// ---------------------------------------------------------------------------
__global__ __launch_bounds__(256) void gemm_msg_upd(
    const _Float16* __restrict__ H, const _Float16* __restrict__ Wm,
    const _Float16* __restrict__ Wu, const float* __restrict__ bu,
    _Float16* __restrict__ m_out, float* __restrict__ agg_out) {
  const int wave = blockIdx.x * (blockDim.x >> 5) + (threadIdx.x >> 5);
  const int lane = threadIdx.x & 31;
  const int row_tile = wave >> 4;
  const int ct       = wave & 15;
  if (row_tile >= NN / 16) return;

  const bool upd = (ct >= 8);
  const int  j0  = (upd ? ct - 8 : ct) * 16;
  const _Float16* W = upd ? Wu : Wm;
  const size_t i0   = (size_t)row_tile * 16;

  v8f acc = wmma_tile(H + i0 * DIM, W + (size_t)j0 * DIM, lane);

  const int n  = lane & 15;
  const int hi = lane >> 4;
  if (upd) {
    const float b = bu[j0 + n];
    float* out = agg_out + i0 * DIM + j0;
#pragma unroll
    for (int r = 0; r < 8; ++r)
      out[(size_t)(r + 8 * hi) * DIM + n] = acc[r] + b;
  } else {
    _Float16* out = m_out + i0 * DIM + j0;
#pragma unroll
    for (int r = 0; r < 8; ++r)
      out[(size_t)(r + 8 * hi) * DIM + n] = (_Float16)acc[r];
  }
}

// Final projection: out = H @ Wout^T + bout (f32 to d_out), 2 column tiles.
__global__ __launch_bounds__(256) void gemm_final(
    const _Float16* __restrict__ H, const _Float16* __restrict__ Wout,
    const float* __restrict__ bout, float* __restrict__ out) {
  const int wave = blockIdx.x * (blockDim.x >> 5) + (threadIdx.x >> 5);
  const int lane = threadIdx.x & 31;
  if (wave >= (NN / 16) * 2) return;
  const int row_tile = wave >> 1;
  const int j0       = (wave & 1) * 16;
  const size_t i0    = (size_t)row_tile * 16;

  v8f acc = wmma_tile(H + i0 * DIM, Wout + (size_t)j0 * DIM, lane);

  const int n  = lane & 15;
  const int hi = lane >> 4;
  const float b = bout[j0 + n];
  float* o = out + i0 * OUTD + j0;
#pragma unroll
  for (int r = 0; r < 8; ++r)
    o[(size_t)(r + 8 * hi) * OUTD + n] = acc[r] + b;
}

// ---------------------------------------------------------------------------
// Edge stage: one wave per edge. Lane l gathers features 4l..4l+3 of m[src]
// (one coalesced 8B f16 load -> 256B/edge, L2-resident), scales by the edge
// weight, and fires four no-return f32 global atomics into L2-resident agg.
// ---------------------------------------------------------------------------
__global__ __launch_bounds__(256) void edge_msg(
    const long long* __restrict__ src, const long long* __restrict__ dst,
    const float* __restrict__ ew, const _Float16* __restrict__ m,
    float* __restrict__ agg) {
  const int idx  = blockIdx.x * blockDim.x + threadIdx.x;
  const int edge = idx >> 5;
  const int lane = idx & 31;
  if (edge >= NE) return;

  const size_t s = (size_t)src[edge];
  const size_t d = (size_t)dst[edge];
  const float  w = ew[edge];

  v4h mv = *(const v4h*)(m + s * DIM + lane * 4);
  float* ap = agg + d * DIM + lane * 4;
  unsafeAtomicAdd(ap + 0, w * (float)mv[0]);
  unsafeAtomicAdd(ap + 1, w * (float)mv[1]);
  unsafeAtomicAdd(ap + 2, w * (float)mv[2]);
  unsafeAtomicAdd(ap + 3, w * (float)mv[3]);
}

// h[i] = tanh(agg[i]) -> packed f16, in place over the h buffer.
__global__ __launch_bounds__(256) void tanh_act(const float* __restrict__ agg,
                                                _Float16* __restrict__ h) {
  const int j = (blockIdx.x * blockDim.x + threadIdx.x) * 2;
  if (j < NN * DIM) {
    float2 v = *(const float2*)(agg + j);
    v2h o;
    o[0] = (_Float16)tanhf(v.x);
    o[1] = (_Float16)tanhf(v.y);
    *(v2h*)(h + j) = o;
  }
}

// Generic fp32 -> f16 conversion (x and the weight matrices), 4 elts/thread.
__global__ __launch_bounds__(256) void cvt_f32_f16(const float* __restrict__ in,
                                                   _Float16* __restrict__ out,
                                                   int n) {
  const int i = (blockIdx.x * blockDim.x + threadIdx.x) * 4;
  if (i >= n) return;
  float4 v = *(const float4*)(in + i);
  v4h o;
  o[0] = (_Float16)v.x; o[1] = (_Float16)v.y;
  o[2] = (_Float16)v.z; o[3] = (_Float16)v.w;
  *(v4h*)(out + i) = o;
}

// ---------------------------------------------------------------------------
extern "C" void kernel_launch(void* const* d_in, const int* in_sizes, int n_in,
                              void* d_out, int out_size, void* d_ws, size_t ws_size,
                              hipStream_t stream) {
  (void)in_sizes; (void)n_in; (void)out_size; (void)ws_size;

  const float*     x  = (const float*)d_in[0];
  const long long* ei = (const long long*)d_in[1];     // int64 [2, E]
  const float*     ew = (const float*)d_in[2];
  const float* Wm[3] = {(const float*)d_in[3], (const float*)d_in[6], (const float*)d_in[9]};
  const float* Wu[3] = {(const float*)d_in[4], (const float*)d_in[7], (const float*)d_in[10]};
  const float* bu[3] = {(const float*)d_in[5], (const float*)d_in[8], (const float*)d_in[11]};
  const float* Wout = (const float*)d_in[12];
  const float* bout = (const float*)d_in[13];

  const size_t nfeat = (size_t)NN * DIM;               // 6.4M
  const size_t wsz   = (size_t)DIM * DIM;              // 16384

  _Float16* h16  = (_Float16*)d_ws;                    // 12.8 MB
  _Float16* m16  = h16 + nfeat;                        // 12.8 MB
  float*    agg  = (float*)(m16 + nfeat);              // 25.6 MB
  _Float16* w16  = (_Float16*)(agg + nfeat);           // 200 KB
  _Float16* Wm16[3], *Wu16[3];
  for (int l = 0; l < 3; ++l) {
    Wm16[l] = w16 + (size_t)l * 2 * wsz;
    Wu16[l] = Wm16[l] + wsz;
  }
  _Float16* Wout16 = w16 + 6 * wsz;

  const long long* src = ei;
  const long long* dst = ei + NE;

  // One-time conversions to f16 workspace (graph-captured, trivial cost).
  cvt_f32_f16<<<(int)(nfeat / 4 / 256), 256, 0, stream>>>(x, h16, (int)nfeat);
  for (int l = 0; l < 3; ++l) {
    cvt_f32_f16<<<16, 256, 0, stream>>>(Wm[l], Wm16[l], (int)wsz);
    cvt_f32_f16<<<16, 256, 0, stream>>>(Wu[l], Wu16[l], (int)wsz);
  }
  cvt_f32_f16<<<4, 256, 0, stream>>>(Wout, Wout16, DIM * OUTD);

  const int gemm_blocks = (NN / 16) * 16 / 8;          // 6250
  const int edge_blocks = NE / 8;                      // 100000 (8 edges/block)
  const int act_blocks  = (int)(nfeat / 2 / 256);      // 12500

  for (int l = 0; l < 3; ++l) {
    gemm_msg_upd<<<gemm_blocks, 256, 0, stream>>>(h16, Wm16[l], Wu16[l], bu[l], m16, agg);
    edge_msg<<<edge_blocks, 256, 0, stream>>>(src, dst, ew, m16, agg);
    tanh_act<<<act_blocks, 256, 0, stream>>>(agg, h16);
  }

  const int fin_blocks = ((NN / 16) * 2 + 7) / 8;      // 782
  gemm_final<<<fin_blocks, 256, 0, stream>>>(h16, Wout16, bout, (float*)d_out);
}
